// GATClassifier_69999376990326
// MI455X (gfx1250) — matrix operand order repaired and checked by reading, not compiled, not run on hardware
//
#include <hip/hip_runtime.h>

typedef __attribute__((ext_vector_type(16))) _Float16 v16h;
typedef __attribute__((ext_vector_type(8)))  _Float16 v8h;
typedef __attribute__((ext_vector_type(8)))  float    v8f;

#define N_NODES   50000
#define N_EDGES   800000
#define IN_DIM    128
#define HID       64
#define HEADS     4
#define HD        256     // HEADS*HID
#define N_CLASSES 10
#define N_GRAPHS  64
#define NEG_SLOPE 0.2f

#define BM   32           // rows per block
#define APAD 40           // halfs per sA row (80B: keeps 16B alignment, spreads banks)
#define BPAD 40           // halfs per sB row (transposed [n][k])

// ---------------------------------------------------------------------------
// WMMA GEMM: C[N x 256] = X[N x K] * W[K x 256], f32 in/out, f16 WMMA core.
// 256 threads = 8 wave32. Block owns 32 rows; wave w owns cols [32w, 32w+32)
// and both 16-row stripes -> 2x2 grid of 16x16 WMMA tiles (4 wmma / K-chunk).
// LDS staging: sA row-major [m][k] (A-frag = two contiguous 8-half runs),
// sB transposed [n][k] (B-frag = two contiguous 8-half runs) so fragments are
// assembled with ds_load_b128 pairs + shufflevector, no scalar half loads.
// Layouts per CDNA5 ISA 7.12.2 (16-bit A 16x32 table; B K-striping 7.12.4).
// ---------------------------------------------------------------------------
__global__ __launch_bounds__(256) void gemm_wmma(const float* __restrict__ X,
                                                 const float* __restrict__ W,
                                                 float* __restrict__ C, int K) {
    __shared__ _Float16 sA[BM * APAD];   // [m][k], padded
    __shared__ _Float16 sB[HD * BPAD];   // [n][k], padded (transposed W chunk)
    const int tid  = threadIdx.x;
    const int wave = tid >> 5;
    const int lane = tid & 31;
    const int hi   = lane >> 4;          // lane half
    const int lm   = lane & 15;          // row (A) / column (B,D) within tile
    const int row0 = blockIdx.x * BM;
    const int col0 = wave * 32;

    v8f acc00 = {}, acc01 = {}, acc10 = {}, acc11 = {};

    for (int k0 = 0; k0 < K; k0 += 32) {
        // ---- stage A (32x32): 4 elems/thread, coalesced reads, clamped rows
        #pragma unroll
        for (int r = 0; r < 4; ++r) {
            int j  = tid + r * 256;
            int am = j >> 5, ak = j & 31;
            int gm = row0 + am; if (gm > N_NODES - 1) gm = N_NODES - 1;
            sA[am * APAD + ak] = (_Float16)X[(size_t)gm * K + k0 + ak];
        }
        // ---- stage B (32x256) transposed to [n][k]: 32 elems/thread
        #pragma unroll
        for (int r = 0; r < 32; ++r) {
            int j  = tid + r * 256;
            int bk = j >> 8, bn = j & 255;
            sB[bn * BPAD + bk] = (_Float16)W[(size_t)(k0 + bk) * HD + bn];
        }
        __syncthreads();

        // ---- A fragments: two contiguous 8-half runs per lane (ds_load_b128 x2)
        v8h a0l = *(const v8h*)&sA[lm * APAD + (hi << 3)];
        v8h a0h = *(const v8h*)&sA[lm * APAD + 16 + (hi << 3)];
        v8h a1l = *(const v8h*)&sA[(lm + 16) * APAD + (hi << 3)];
        v8h a1h = *(const v8h*)&sA[(lm + 16) * APAD + 16 + (hi << 3)];
        v16h A0 = __builtin_shufflevector(a0l, a0h, 0,1,2,3,4,5,6,7,8,9,10,11,12,13,14,15);
        v16h A1 = __builtin_shufflevector(a1l, a1h, 0,1,2,3,4,5,6,7,8,9,10,11,12,13,14,15);

        // ---- B fragments: two 8-half runs per lane (ds_load_b128 x2)
        v8h b0l = *(const v8h*)&sB[(col0 + lm) * BPAD + (hi << 3)];
        v8h b0h = *(const v8h*)&sB[(col0 + lm) * BPAD + 16 + (hi << 3)];
        v8h b1l = *(const v8h*)&sB[(col0 + 16 + lm) * BPAD + (hi << 3)];
        v8h b1h = *(const v8h*)&sB[(col0 + 16 + lm) * BPAD + 16 + (hi << 3)];
        v16h B0 = __builtin_shufflevector(b0l, b0h, 0,1,2,3,4,5,6,7,8,9,10,11,12,13,14,15);
        v16h B1 = __builtin_shufflevector(b1l, b1h, 0,1,2,3,4,5,6,7,8,9,10,11,12,13,14,15);

        acc00 = __builtin_amdgcn_wmma_f32_16x16x32_f16(false, A0, false, B0,
                                                       (short)0, acc00, false, false);
        acc01 = __builtin_amdgcn_wmma_f32_16x16x32_f16(false, A0, false, B1,
                                                       (short)0, acc01, false, false);
        acc10 = __builtin_amdgcn_wmma_f32_16x16x32_f16(false, A1, false, B0,
                                                       (short)0, acc10, false, false);
        acc11 = __builtin_amdgcn_wmma_f32_16x16x32_f16(false, A1, false, B1,
                                                       (short)0, acc11, false, false);
        __syncthreads();
    }

    // ---- store D per 32-bit C/D layout (lane half selects M 0-7 vs 8-15)
    #pragma unroll
    for (int r = 0; r < 8; ++r) {
        int row = (hi << 3) + r;
        if (row0 + row < N_NODES) {
            C[(size_t)(row0 + row) * HD + col0 + lm]      = acc00[r];
            C[(size_t)(row0 + row) * HD + col0 + 16 + lm] = acc01[r];
        }
        if (row0 + 16 + row < N_NODES) {
            C[(size_t)(row0 + 16 + row) * HD + col0 + lm]      = acc10[r];
            C[(size_t)(row0 + 16 + row) * HD + col0 + 16 + lm] = acc11[r];
        }
    }
}

// ---------------------------------------------------------------------------
__device__ __forceinline__ float atomicMaxF(float* addr, float val) {
    unsigned int* ua  = (unsigned int*)addr;
    unsigned int  old = *ua;
    while (true) {
        float f = __uint_as_float(old);
        if (f >= val) break;
        unsigned int assumed = old;
        old = atomicCAS(ua, assumed, __float_as_uint(val));
        if (old == assumed) break;
    }
    return __uint_as_float(old);
}

__global__ void fill_f32(float* __restrict__ p, float v, int n) {
    int t = blockIdx.x * blockDim.x + threadIdx.x;
    if (t < n) p[t] = v;
}

// el[n,h] = <feat[n,h,:], al[h,:]>, er likewise
__global__ void node_attn(const float* __restrict__ feat, const float* __restrict__ al,
                          const float* __restrict__ ar, float* __restrict__ el,
                          float* __restrict__ er) {
    int t = blockIdx.x * blockDim.x + threadIdx.x;
    if (t >= N_NODES * HEADS) return;
    int n = t >> 2, h = t & 3;
    const float* f = feat + (size_t)n * HD + h * HID;
    const float* a = al + h * HID;
    const float* b = ar + h * HID;
    float sl = 0.f, sr = 0.f;
    #pragma unroll 8
    for (int d = 0; d < HID; ++d) { float v = f[d]; sl += v * a[d]; sr += v * b[d]; }
    el[t] = sl; er[t] = sr;
}

// pass 1: e = leaky_relu(el[src]+er[dst]); segment max over dst via atomic max
__global__ void edge_attn_max(const int* __restrict__ ei, const float* __restrict__ el,
                              const float* __restrict__ er, float* __restrict__ ebuf,
                              float* __restrict__ nmax) {
    int t = blockIdx.x * blockDim.x + threadIdx.x;
    if (t >= N_EDGES * HEADS) return;
    int e = t >> 2, h = t & 3;
    int s = ei[e], d = ei[N_EDGES + e];
    float v = el[s * 4 + h] + er[d * 4 + h];
    v = v > 0.f ? v : NEG_SLOPE * v;
    ebuf[t] = v;
    atomicMaxF(&nmax[d * 4 + h], v);
}

// pass 2: ee = exp(e - max[dst]); segment sum over dst
__global__ void edge_exp_sum(const int* __restrict__ ei, float* __restrict__ ebuf,
                             const float* __restrict__ nmax, float* __restrict__ den) {
    int t = blockIdx.x * blockDim.x + threadIdx.x;
    if (t >= N_EDGES * HEADS) return;
    int e = t >> 2, h = t & 3;
    int d = ei[N_EDGES + e];
    float ee = __expf(ebuf[t] - nmax[d * 4 + h]);
    ebuf[t] = ee;
    atomicAdd(&den[d * 4 + h], ee);
}

// pass 3: one wave per edge. lane owns 8 contiguous dims (single head each):
// rst[dst] += alpha * feat[src]
__global__ __launch_bounds__(256) void edge_aggregate(const int* __restrict__ ei,
        const float* __restrict__ ebuf, const float* __restrict__ den,
        const float* __restrict__ feat, float* __restrict__ rst) {
    int wave = threadIdx.x >> 5, lane = threadIdx.x & 31;
    int e = blockIdx.x * 8 + wave;
    if (e >= N_EDGES) return;
    int s = ei[e], d = ei[N_EDGES + e];
    __builtin_prefetch(feat + (size_t)s * HD + lane * 8, 0, 0);  // global_prefetch_b8
    int h = lane >> 3;
    float alpha = ebuf[e * 4 + h] / fmaxf(den[d * 4 + h], 1e-9f);
    int base = lane * 8;
    const float4 f0 = *reinterpret_cast<const float4*>(feat + (size_t)s * HD + base);
    const float4 f1 = *reinterpret_cast<const float4*>(feat + (size_t)s * HD + base + 4);
    float* rp = rst + (size_t)d * HD + base;
    atomicAdd(rp + 0, alpha * f0.x); atomicAdd(rp + 1, alpha * f0.y);
    atomicAdd(rp + 2, alpha * f0.z); atomicAdd(rp + 3, alpha * f0.w);
    atomicAdd(rp + 4, alpha * f1.x); atomicAdd(rp + 5, alpha * f1.y);
    atomicAdd(rp + 6, alpha * f1.z); atomicAdd(rp + 7, alpha * f1.w);
}

__global__ void bias_relu(float* __restrict__ x, const float* __restrict__ b, int n) {
    int t = blockIdx.x * blockDim.x + threadIdx.x;
    if (t < n) x[t] = fmaxf(x[t] + b[t & 255], 0.f);
}

__global__ void head_mean(const float* __restrict__ h2, float* __restrict__ hm) {
    int t = blockIdx.x * blockDim.x + threadIdx.x;
    if (t >= N_NODES * HID) return;
    int n = t >> 6, d = t & 63;
    const float* p = h2 + (size_t)n * HD + d;
    hm[t] = 0.25f * (p[0] + p[64] + p[128] + p[192]);
}

__global__ void graph_pool(const float* __restrict__ hm, const int* __restrict__ gid,
                           float* __restrict__ pooled) {
    int t = blockIdx.x * blockDim.x + threadIdx.x;
    if (t >= N_NODES * HID) return;
    int n = t >> 6, d = t & 63;
    atomicMaxF(&pooled[gid[n] * HID + d], hm[t]);
}

__global__ void classify(const float* __restrict__ pooled, const float* __restrict__ Wc,
                         const float* __restrict__ bc, float* __restrict__ out) {
    int t = blockIdx.x * blockDim.x + threadIdx.x;
    if (t >= N_GRAPHS * N_CLASSES) return;
    int g = t / N_CLASSES, c = t % N_CLASSES;
    float s = bc[c];
    #pragma unroll 8
    for (int d = 0; d < HID; ++d) s += pooled[g * HID + d] * Wc[d * N_CLASSES + c];
    out[t] = s;
}

// ---------------------------------------------------------------------------
extern "C" void kernel_launch(void* const* d_in, const int* in_sizes, int n_in,
                              void* d_out, int out_size, void* d_ws, size_t ws_size,
                              hipStream_t stream) {
    (void)in_sizes; (void)n_in; (void)out_size; (void)ws_size;
    const float* h   = (const float*)d_in[0];
    const float* W1  = (const float*)d_in[1];
    const float* al1 = (const float*)d_in[2];
    const float* ar1 = (const float*)d_in[3];
    const float* b1  = (const float*)d_in[4];
    const float* W2  = (const float*)d_in[5];
    const float* al2 = (const float*)d_in[6];
    const float* ar2 = (const float*)d_in[7];
    const float* b2  = (const float*)d_in[8];
    const float* Wc  = (const float*)d_in[9];
    const float* bc  = (const float*)d_in[10];
    const int*   ei  = (const int*)d_in[11];
    const int*   gid = (const int*)d_in[12];
    float* out = (float*)d_out;

    float* ws = (float*)d_ws;
    size_t off = 0;
    float* feat   = ws + off; off += (size_t)N_NODES * HD;     // projections
    float* rbuf   = ws + off; off += (size_t)N_NODES * HD;     // rst / h1 / h2
    float* ebuf   = ws + off; off += (size_t)N_EDGES * HEADS;  // e -> ee
    float* el     = ws + off; off += (size_t)N_NODES * HEADS;
    float* er     = ws + off; off += (size_t)N_NODES * HEADS;
    float* nmax   = ws + off; off += (size_t)N_NODES * HEADS;
    float* den    = ws + off; off += (size_t)N_NODES * HEADS;
    float* hm     = ws + off; off += (size_t)N_NODES * HID;
    float* pooled = ws + off; off += (size_t)N_GRAPHS * HID;

    const dim3 B(256);
    const dim3 GEMM_G((N_NODES + BM - 1) / BM);
    auto nb = [](long n) { return dim3((unsigned)((n + 255) / 256)); };

    // ===================== Layer 1 =====================
    gemm_wmma<<<GEMM_G, B, 0, stream>>>(h, W1, feat, IN_DIM);
    node_attn<<<nb(N_NODES * HEADS), B, 0, stream>>>(feat, al1, ar1, el, er);
    fill_f32<<<nb(N_NODES * HEADS), B, 0, stream>>>(nmax, -3.0e38f, N_NODES * HEADS);
    fill_f32<<<nb(N_NODES * HEADS), B, 0, stream>>>(den, 0.f, N_NODES * HEADS);
    fill_f32<<<nb((long)N_NODES * HD), B, 0, stream>>>(rbuf, 0.f, N_NODES * HD);
    edge_attn_max<<<nb((long)N_EDGES * HEADS), B, 0, stream>>>(ei, el, er, ebuf, nmax);
    edge_exp_sum<<<nb((long)N_EDGES * HEADS), B, 0, stream>>>(ei, ebuf, nmax, den);
    edge_aggregate<<<dim3(N_EDGES / 8), B, 0, stream>>>(ei, ebuf, den, feat, rbuf);
    bias_relu<<<nb((long)N_NODES * HD), B, 0, stream>>>(rbuf, b1, N_NODES * HD);  // h1 in rbuf

    // ===================== Layer 2 =====================
    gemm_wmma<<<GEMM_G, B, 0, stream>>>(rbuf, W2, feat, HD);
    node_attn<<<nb(N_NODES * HEADS), B, 0, stream>>>(feat, al2, ar2, el, er);
    fill_f32<<<nb(N_NODES * HEADS), B, 0, stream>>>(nmax, -3.0e38f, N_NODES * HEADS);
    fill_f32<<<nb(N_NODES * HEADS), B, 0, stream>>>(den, 0.f, N_NODES * HEADS);
    fill_f32<<<nb((long)N_NODES * HD), B, 0, stream>>>(rbuf, 0.f, N_NODES * HD);  // after gemm2 (stream order)
    edge_attn_max<<<nb((long)N_EDGES * HEADS), B, 0, stream>>>(ei, el, er, ebuf, nmax);
    edge_exp_sum<<<nb((long)N_EDGES * HEADS), B, 0, stream>>>(ei, ebuf, nmax, den);
    edge_aggregate<<<dim3(N_EDGES / 8), B, 0, stream>>>(ei, ebuf, den, feat, rbuf);
    bias_relu<<<nb((long)N_NODES * HD), B, 0, stream>>>(rbuf, b2, N_NODES * HD);  // h2 in rbuf

    // ===================== Readout =====================
    head_mean<<<nb((long)N_NODES * HID), B, 0, stream>>>(rbuf, hm);
    fill_f32<<<nb(N_GRAPHS * HID), B, 0, stream>>>(pooled, -3.0e38f, N_GRAPHS * HID);
    graph_pool<<<nb((long)N_NODES * HID), B, 0, stream>>>(hm, gid, pooled);
    classify<<<nb(N_GRAPHS * N_CLASSES), B, 0, stream>>>(pooled, Wc, bc, out);
}
